// GRU_8211977470410
// MI455X (gfx1250) — matrix operand order repaired and checked by reading, not compiled, not run on hardware
//
#include <hip/hip_runtime.h>
#include <hip/hip_fp16.h>
#include <stdint.h>

typedef __attribute__((ext_vector_type(16))) _Float16 v16h;
typedef __attribute__((ext_vector_type(8)))  float    v8f;
typedef __attribute__((ext_vector_type(2)))  float    v2f;
typedef __attribute__((ext_vector_type(8)))  unsigned v8u;

#define TLEN 512
#define HID  32

#if defined(__has_builtin)
#  if __has_builtin(__builtin_amdgcn_wmma_f32_16x16x4_f32)
#    define HAVE_WMMA_F32X4 1
#  endif
#  if __has_builtin(__builtin_amdgcn_tanhf)
#    define HAVE_TANH 1
#  endif
#endif

__device__ __forceinline__ float ftanh(float x) {
#ifdef HAVE_TANH
  return __builtin_amdgcn_tanhf(x);             // v_tanh_f32 (gfx1250)
#else
  float e = __builtin_amdgcn_exp2f(2.8853900817779268f * x);
  return 1.0f - 2.0f * __builtin_amdgcn_rcpf(e + 1.0f);
#endif
}

__device__ __forceinline__ float fsigmoid(float x) {
#ifdef HAVE_TANH
  // 3-op chain, single transcendental: sigmoid(x) = 0.5 + 0.5*tanh(x/2)
  return fmaf(0.5f, __builtin_amdgcn_tanhf(0.5f * x), 0.5f);
#else
  float e = __builtin_amdgcn_exp2f(-1.4426950408889634f * x);
  return __builtin_amdgcn_rcpf(1.0f + e);
#endif
}

__device__ __forceinline__ unsigned pk16(float lo, float hi) {
  return __builtin_bit_cast(unsigned, __builtin_amdgcn_cvt_pkrtz(lo, hi));
}

// One wave32 per 16-row batch tile. Forward GRU: 512 steps, W_hh stationary in
// registers as six 16x32 f16 A-fragments; h lives as packed-f16 B-fragment,
// rebuilt each step from the f32 D-layout result via a lane^16 half-exchange.
// x is consumed as one float4 per 4 steps, pipelined one chunk ahead so the
// load wait never sits on the recurrence's critical path.
// Backward GRU: exactly one step from h0=0 (W_hh_b unused). MLP head fused.
__global__ __launch_bounds__(32) void gru_fused_kernel(
    const float* __restrict__ x,
    const float* __restrict__ Wih_f, const float* __restrict__ Whh_f,
    const float* __restrict__ bih_f, const float* __restrict__ bhh_f,
    const float* __restrict__ Wih_b,
    const float* __restrict__ bih_b, const float* __restrict__ bhh_b,
    const float* __restrict__ W1, const float* __restrict__ b1,
    const float* __restrict__ W2, const float* __restrict__ b2,
    float* __restrict__ out)
{
  const int lane = threadIdx.x & 31;
  const int m    = lane & 15;   // batch column within tile (B/C/D lane layout)
  const int hi   = lane >> 4;   // half-wave selector
  const int b0   = blockIdx.x * 16;
  const int b    = b0 + m;

  __shared__ float hcat[16][68];   // [batch][64 features], padded stride

  // ---- W_hh_f as six 16x32 f16 A-fragments (row M = gate row, K = hidden) ----
  // A k order used here (contiguous, matches hB): lane m: elems 0..15 -> K = e + 16*hi.
  v16h afrag[6];
#pragma unroll
  for (int t = 0; t < 6; ++t) {
#pragma unroll
    for (int e = 0; e < 16; ++e) {
      int k = e + 16 * hi;
      afrag[t][e] = (_Float16)Whh_f[(16 * t + m) * HID + k];
    }
  }

  // ---- bias C-inits (D layout: row = vgpr + 8*hi + 16*tile, col = m) ----
  float c_rz[4][8];                 // r,z tiles: b_hh + b_ih folded together
#pragma unroll
  for (int t = 0; t < 4; ++t)
#pragma unroll
    for (int v = 0; v < 8; ++v) {
      int row = 16 * t + v + 8 * hi;
      c_rz[t][v] = bhh_f[row] + bih_f[row];
    }
  float c_nh[2][8], c_nx[2][8];     // n gate: b_hh stays inside r*(.), b_ih outside
#pragma unroll
  for (int i = 0; i < 2; ++i)
#pragma unroll
    for (int v = 0; v < 8; ++v) {
      int row = 64 + 16 * i + v + 8 * hi;
      c_nh[i][v] = bhh_f[row];
      c_nx[i][v] = bih_f[row];
    }

  const float xmask = hi ? 0.0f : 1.0f;   // x B-fragment: only K=0 row non-zero

#ifdef HAVE_WMMA_F32X4
  // W_ih_f as 16x4 f32 A-fragments (only K=0 column non-zero, input dim = 1)
  v2f aih[6];
#pragma unroll
  for (int t = 0; t < 6; ++t) {
    aih[t].x = xmask * Wih_f[16 * t + m];
    aih[t].y = 0.0f;
  }
#else
  float wih[6][8];
#pragma unroll
  for (int t = 0; t < 6; ++t)
#pragma unroll
    for (int v = 0; v < 8; ++v)
      wih[t][v] = Wih_f[16 * t + v + 8 * hi];
#endif

  // ---- recurrent state: f32 in D layout + packed-f16 B-fragment ----
  float h1[8], h2[8];
#pragma unroll
  for (int v = 0; v < 8; ++v) { h1[v] = 0.0f; h2[v] = 0.0f; }
  v8u hBu = {0u, 0u, 0u, 0u, 0u, 0u, 0u, 0u};

  const float* xrow = x + (size_t)b * TLEN;
  const float4* xq  = (const float4*)xrow;

  // ---- one GRU step (everything register-resident) ----
  auto gru_step = [&](float xv) {
    v16h hB = __builtin_bit_cast(v16h, hBu);
    v8f drz[4], dnx[2], dnh[2];
#ifdef HAVE_WMMA_F32X4
    v2f xB; xB.x = xv * xmask; xB.y = 0.0f;       // 4x16 f32 B, only K=0 row
#pragma unroll
    for (int tt = 0; tt < 4; ++tt) {
      v8f c;
#pragma unroll
      for (int v = 0; v < 8; ++v) c[v] = c_rz[tt][v];
      c = __builtin_amdgcn_wmma_f32_16x16x4_f32(false, aih[tt], false, xB,
                                                (short)0, c, false, false);
      drz[tt] = __builtin_amdgcn_wmma_f32_16x16x32_f16(false, afrag[tt], false, hB,
                                                       (short)0, c, false, false);
    }
#pragma unroll
    for (int i = 0; i < 2; ++i) {
      v8f cx, ch;
#pragma unroll
      for (int v = 0; v < 8; ++v) { cx[v] = c_nx[i][v]; ch[v] = c_nh[i][v]; }
      dnx[i] = __builtin_amdgcn_wmma_f32_16x16x4_f32(false, aih[4 + i], false, xB,
                                                     (short)0, cx, false, false);
      dnh[i] = __builtin_amdgcn_wmma_f32_16x16x32_f16(false, afrag[4 + i], false, hB,
                                                      (short)0, ch, false, false);
    }
#else
#pragma unroll
    for (int tt = 0; tt < 4; ++tt) {
      v8f c;
#pragma unroll
      for (int v = 0; v < 8; ++v) c[v] = fmaf(xv, wih[tt][v], c_rz[tt][v]);
      drz[tt] = __builtin_amdgcn_wmma_f32_16x16x32_f16(false, afrag[tt], false, hB,
                                                       (short)0, c, false, false);
    }
#pragma unroll
    for (int i = 0; i < 2; ++i) {
      v8f cx, ch;
#pragma unroll
      for (int v = 0; v < 8; ++v) {
        cx[v] = fmaf(xv, wih[4 + i][v], c_nx[i][v]);
        ch[v] = c_nh[i][v];
      }
      dnx[i] = cx;
      dnh[i] = __builtin_amdgcn_wmma_f32_16x16x32_f16(false, afrag[4 + i], false, hB,
                                                      (short)0, ch, false, false);
    }
#endif

    // gates + state update (per lane: batch col fixed, 16 hidden rows)
#pragma unroll
    for (int i = 0; i < 2; ++i) {
      float* h = i ? h2 : h1;
#pragma unroll
      for (int v = 0; v < 8; ++v) {
        float r = fsigmoid(drz[i][v]);
        float z = fsigmoid(drz[2 + i][v]);
        float n = ftanh(dnx[i][v] + r * dnh[i][v]);
        h[v] = (1.0f - z) * n + z * h[v];
      }
    }

    // rebuild B-fragment of h: pack to f16, lane^16 half-exchange.
    // low lane holds hid 0..7 (h1) & 16..23 (h2); high lane 8..15 & 24..31.
    // B here uses low lanes K=0..15, high lanes K=16..31 (same order as afrag).
#pragma unroll
    for (int p = 0; p < 4; ++p) {
      unsigned own1 = pk16(h1[2 * p], h1[2 * p + 1]);
      unsigned own2 = pk16(h2[2 * p], h2[2 * p + 1]);
      unsigned send = hi ? own1 : own2;
      unsigned rcv  = (unsigned)__shfl_xor((int)send, 16, 32);
      hBu[p]     = hi ? rcv  : own1;
      hBu[4 + p] = hi ? own2 : rcv;
    }
  };

  // ---- main recurrence, x pipelined one float4 chunk ahead ----
  float4 xcur = xq[0];
  for (int i = 0; i < TLEN / 4; ++i) {
    float4 xnext = xq[(i + 1) & (TLEN / 4 - 1)];          // branchless wrap
    __builtin_prefetch(xq + ((i + 8) & (TLEN / 4 - 1)), 0, 0);
    gru_step(xcur.x);
    gru_step(xcur.y);
    gru_step(xcur.z);
    gru_step(xcur.w);
    xcur = xnext;
  }

  // ---- stage forward hidden into LDS feature block ----
#pragma unroll
  for (int v = 0; v < 8; ++v) {
    hcat[m][v + 8 * hi]      = h1[v];
    hcat[m][16 + v + 8 * hi] = h2[v];
  }

  // ---- backward direction: one step from h0 = 0 consuming x[T-1] ----
  {
    float xb = xrow[TLEN - 1];
#pragma unroll
    for (int i = 0; i < 2; ++i)
#pragma unroll
      for (int v = 0; v < 8; ++v) {
        int hid = 16 * i + v + 8 * hi;
        float r = fsigmoid(xb * Wih_b[hid]      + bih_b[hid]      + bhh_b[hid]);
        float z = fsigmoid(xb * Wih_b[32 + hid] + bih_b[32 + hid] + bhh_b[32 + hid]);
        float n = ftanh   (xb * Wih_b[64 + hid] + bih_b[64 + hid] + r * bhh_b[64 + hid]);
        hcat[m][32 + hid] = (1.0f - z) * n;      // z * h0 == 0
      }
  }
  __syncthreads();

  // ---- MLP head: Linear(64,16) -> ReLU -> Linear(16,1) -> Sigmoid ----
  if (lane < 16) {
    float acc = b2[0];
#pragma unroll 4
    for (int k = 0; k < 16; ++k) {
      float a = b1[k];
#pragma unroll 8
      for (int j = 0; j < 64; ++j) a = fmaf(hcat[lane][j], W1[k * 64 + j], a);
      a = fmaxf(a, 0.0f);
      acc = fmaf(a, W2[k], acc);
    }
    out[b0 + lane] = fsigmoid(acc);
  }
}

extern "C" void kernel_launch(void* const* d_in, const int* in_sizes, int n_in,
                              void* d_out, int out_size, void* d_ws, size_t ws_size,
                              hipStream_t stream) {
  (void)n_in; (void)out_size; (void)d_ws; (void)ws_size;
  const float* x     = (const float*)d_in[0];
  const float* Wih_f = (const float*)d_in[1];
  const float* Whh_f = (const float*)d_in[2];
  const float* bih_f = (const float*)d_in[3];
  const float* bhh_f = (const float*)d_in[4];
  const float* Wih_b = (const float*)d_in[5];
  // d_in[6] = W_hh_b: mathematically unused (backward output is a single step
  // from h0 == 0, so h @ W_hh_b.T == 0).
  const float* bih_b = (const float*)d_in[7];
  const float* bhh_b = (const float*)d_in[8];
  const float* W1    = (const float*)d_in[9];
  const float* b1    = (const float*)d_in[10];
  const float* W2    = (const float*)d_in[11];
  const float* b2    = (const float*)d_in[12];

  int B = in_sizes[0] / TLEN;          // 2048
  dim3 grid(B / 16), block(32);        // one wave32 per 16-row batch tile
  gru_fused_kernel<<<grid, block, 0, stream>>>(
      x, Wih_f, Whh_f, bih_f, bhh_f, Wih_b, bih_b, bhh_b,
      W1, b1, W2, b2, (float*)d_out);
}
